// MiniMoeBackboneModel_73796128080301
// MI455X (gfx1250) — matrix operand order repaired and checked by reading, not compiled, not run on hardware
//
#include <hip/hip_runtime.h>
#include <hip/hip_bf16.h>
#include <math.h>

// ---------------- model dims ----------------
#define D_MODEL 1024
#define SEQ     1024
#define BATCH   2
#define NTOK    (BATCH*SEQ)      // 2048
#define NHEAD   16
#define DHEAD   64
#define FFN_DIM 4096
#define NEXP    8
#define WIN     256
#define VOCAB   32000

typedef __bf16 bf16_t;
typedef __attribute__((ext_vector_type(16))) __bf16 v16bf;
typedef __attribute__((ext_vector_type(8)))  float  v8f;

__device__ __forceinline__ bf16_t f2bf(float f) {
  unsigned u = __builtin_bit_cast(unsigned, f);
  unsigned r = u + 0x7FFFu + ((u >> 16) & 1u);   // round-to-nearest-even
  unsigned short h = (unsigned short)(r >> 16);
  return __builtin_bit_cast(bf16_t, h);
}

__device__ __forceinline__ v16bf pack_bf16(float4 a0, float4 a1, float4 a2, float4 a3) {
  v16bf r;
  r[0]=f2bf(a0.x);  r[1]=f2bf(a0.y);  r[2]=f2bf(a0.z);  r[3]=f2bf(a0.w);
  r[4]=f2bf(a1.x);  r[5]=f2bf(a1.y);  r[6]=f2bf(a1.z);  r[7]=f2bf(a1.w);
  r[8]=f2bf(a2.x);  r[9]=f2bf(a2.y);  r[10]=f2bf(a2.z); r[11]=f2bf(a2.w);
  r[12]=f2bf(a3.x); r[13]=f2bf(a3.y); r[14]=f2bf(a3.z); r[15]=f2bf(a3.w);
  return r;
}

// ---------------- GEMM epilogue modes ----------------
#define GEMM_PLAIN      0   // C = A@W + bias
#define GEMM_RELU       1   // C = relu(A@W + bias)
#define GEMM_RES        2   // C = res + A@W + bias
#define GEMM_SCALE_ACC  3   // C += rowscale[m] * (A@W + bias)

// LDS tile geometry (floats).  Non-trans: [k][n] rows of 64 padded to 68
// (272B, 16B aligned for async b128 writes).  Trans: [n][k] rows of 32
// padded to 36 (144B).  Double buffered.
#define LDSA_STRIDE 68
#define LDSB_STRIDE 36
#define LDS_BUF_F   2304     // >= max(32*68, 64*36) = 2304

// C[M,N] = f(A[M,K] @ W + bias), bf16 WMMA with fp32 accumulation.
// W element (k,n) = WT ? W[n*ldw + k] : W[k*ldw + n].
// Block: 256 threads = 8 waves. Tile: 256(M) x 64(N); each wave owns 32(M)x64(N)
// (two A fragments sharing each B fragment -> 8 WMMAs per 32-wide K step).
// Weight tiles are DMA'd to LDS with GLOBAL_LOAD_ASYNC_TO_LDS_B128 and
// double-buffered so the async engine overlaps the WMMAs (ASYNCcnt pipelining).
template <int WT>
__global__ __launch_bounds__(256)
void gemm_bf16_wmma(const float* __restrict__ A, int lda,
                    const float* __restrict__ W, int ldw,
                    const float* __restrict__ bias,
                    const float* __restrict__ res,
                    const float* __restrict__ rowscale, int rs_stride,
                    float* __restrict__ C, int ldc,
                    int M, int N, int Kdim, int mode)
{
  __shared__ __align__(16) float Bs[2][LDS_BUF_F];

  const int tid   = threadIdx.x;
  const int lane  = tid & 31;
  const int wv    = tid >> 5;                 // wave id 0..7 -> M sub-tile
  const int n0    = blockIdx.x * 64;
  const int m0    = blockIdx.y * 256;
  const int kb    = (lane >> 4) * 16;         // A fragment k base (contiguous k = 16h + s)
  const int khalf = (lane >> 4) * 16;         // B fragment k base

  v8f acc0[4] = {{}, {}, {}, {}};             // rows m0+wv*32    .. +15
  v8f acc1[4] = {{}, {}, {}, {}};             // rows m0+wv*32+16 .. +31

  const float* Arow0 = A + (size_t)(m0 + wv * 32      + (lane & 15)) * lda;
  const float* Arow1 = A + (size_t)(m0 + wv * 32 + 16 + (lane & 15)) * lda;
  const int KT = Kdim >> 5;                   // 32-wide K tiles

  // Issue the async DMA of one 32(K) x 64(N) weight tile into LDS buffer `buf`.
  // 512 16B chunks total = 8 waves x 2 issues x 32 lanes.
  auto stage = [&](int kt, int buf) {
    const int k0 = kt * 32;
    const unsigned lbase = (unsigned)(size_t)(&Bs[buf][0]);
    #pragma unroll
    for (int i = 0; i < 2; ++i) {
      const int c = wv * 64 + i * 32 + lane;
      const float* g; unsigned l;
      if (WT) {                   // [n][k] layout: runs contiguous along k
        const int nn = c >> 3, j = c & 7;
        g = W + (size_t)(n0 + nn) * ldw + k0 + 4 * j;
        l = lbase + (unsigned)(nn * (LDSB_STRIDE * 4) + j * 16);
      } else {                    // [k][n] layout: runs contiguous along n
        const int kk = c >> 4, j = c & 15;
        g = W + (size_t)(k0 + kk) * ldw + n0 + 4 * j;
        l = lbase + (unsigned)(kk * (LDSA_STRIDE * 4) + j * 16);
      }
      asm volatile("global_load_async_to_lds_b128 %0, %1, off"
                   :: "v"(l), "v"((unsigned long long)(size_t)g) : "memory");
    }
  };

  stage(0, 0);

  for (int kt = 0; kt < KT; ++kt) {
    const int cur = kt & 1;
    if (kt + 1 < KT) {
      stage(kt + 1, cur ^ 1);     // overlap next tile's DMA with this compute
      asm volatile("s_wait_asynccnt 0x2" ::: "memory");  // tile kt landed
    } else {
      asm volatile("s_wait_asynccnt 0x0" ::: "memory");
    }
    __syncthreads();              // all waves' copies complete -> tile valid

    const int k0 = kt * 32;
    if (k0 + 32 < Kdim) {
      __builtin_prefetch(Arow0 + k0 + 32, 0, 1);
      __builtin_prefetch(Arow1 + k0 + 32, 0, 1);
    }

    // ---- A fragments: lane holds one row, two contiguous 8-float k runs ----
    const float4* p00 = (const float4*)(Arow0 + k0 + kb);
    const float4* p01 = (const float4*)(Arow0 + k0 + kb + 8);
    v16bf af0 = pack_bf16(p00[0], p00[1], p01[0], p01[1]);
    const float4* p10 = (const float4*)(Arow1 + k0 + kb);
    const float4* p11 = (const float4*)(Arow1 + k0 + kb + 8);
    v16bf af1 = pack_bf16(p10[0], p10[1], p11[0], p11[1]);

    const float* Bb = &Bs[cur][0];

    // ---- 8 WMMAs: wave covers 32(M) x 64(N); B fragment reused twice ----
    #pragma unroll
    for (int f = 0; f < 4; ++f) {
      const int nf = f * 16 + (lane & 15);
      v16bf bfr;
      if (WT) {                   // contiguous along k: four float4 LDS loads
        const float4* pb = (const float4*)(Bb + nf * LDSB_STRIDE + khalf);
        bfr = pack_bf16(pb[0], pb[1], pb[2], pb[3]);
      } else {                    // strided column reads
        #pragma unroll
        for (int e = 0; e < 16; ++e)
          bfr[e] = f2bf(Bb[(khalf + e) * LDSA_STRIDE + nf]);
      }
      acc0[f] = __builtin_amdgcn_wmma_f32_16x16x32_bf16(
                  false, af0, false, bfr, (short)0, acc0[f], false, false);
      acc1[f] = __builtin_amdgcn_wmma_f32_16x16x32_bf16(
                  false, af1, false, bfr, (short)0, acc1[f], false, false);
    }
    __syncthreads();              // all reads done before buffer is re-DMA'd
  }

  // ---- epilogue (C/D layout: m = (lane>=16?8:0)+r, n = lane&15) ----
  #pragma unroll
  for (int half = 0; half < 2; ++half) {
    const int mbase = m0 + wv * 32 + half * 16 + ((lane >> 4) * 8);
    #pragma unroll
    for (int f = 0; f < 4; ++f) {
      const int n = n0 + f * 16 + (lane & 15);
      if (n >= N) continue;
      #pragma unroll
      for (int r = 0; r < 8; ++r) {
        const int m = mbase + r;
        if (m >= M) continue;
        float vv = half ? acc1[f][r] : acc0[f][r];
        if (bias) vv += bias[n];
        if (mode == GEMM_RELU)            vv = vv > 0.f ? vv : 0.f;
        else if (mode == GEMM_RES)        vv += res[(size_t)m * ldc + n];
        else if (mode == GEMM_SCALE_ACC)  vv = C[(size_t)m * ldc + n]
                                             + rowscale[(size_t)m * rs_stride] * vv;
        C[(size_t)m * ldc + n] = vv;
      }
    }
  }
}

// ---------------- embedding gather ----------------
__global__ __launch_bounds__(256)
void embed_kernel(const int* __restrict__ ids, const float* __restrict__ emb,
                  float* __restrict__ h)
{
  int t = blockIdx.x;
  size_t src = (size_t)ids[t] * D_MODEL;
  for (int d = threadIdx.x; d < D_MODEL; d += 256)
    h[(size_t)t * D_MODEL + d] = emb[src + d];
}

// ---------------- layernorm ----------------
__global__ __launch_bounds__(256)
void layernorm_kernel(const float* __restrict__ x, const float* __restrict__ s,
                      const float* __restrict__ b, float* __restrict__ y)
{
  __shared__ float rsum[256], rsq[256];
  int t = blockIdx.x, tid = threadIdx.x;
  const float* row = x + (size_t)t * D_MODEL;
  float sum = 0.f, sq = 0.f;
  for (int d = tid; d < D_MODEL; d += 256) { float v = row[d]; sum += v; sq += v * v; }
  rsum[tid] = sum; rsq[tid] = sq;
  __syncthreads();
  for (int off = 128; off > 0; off >>= 1) {
    if (tid < off) { rsum[tid] += rsum[tid + off]; rsq[tid] += rsq[tid + off]; }
    __syncthreads();
  }
  float mu  = rsum[0] * (1.f / D_MODEL);
  float var = rsq[0] * (1.f / D_MODEL) - mu * mu;
  float inv = rsqrtf(var + 1e-5f);
  for (int d = tid; d < D_MODEL; d += 256)
    y[(size_t)t * D_MODEL + d] = (row[d] - mu) * inv * s[d] + b[d];
}

// ---------------- windowed causal attention (per (b,h,q) block) ----------------
__global__ __launch_bounds__(64)
void attention_kernel(const float* __restrict__ q, const float* __restrict__ k,
                      const float* __restrict__ v, float* __restrict__ ctx)
{
  __shared__ float qs[DHEAD];
  __shared__ float sc[WIN];
  __shared__ float red[64];
  int qpos = blockIdx.x;
  int bh   = blockIdx.y;
  int b    = bh / NHEAD, hh = bh % NHEAD;
  int t    = b * SEQ + qpos;
  int js   = qpos - (WIN - 1); if (js < 0) js = 0;
  int cnt  = qpos - js + 1;
  int tid  = threadIdx.x;

  qs[tid] = q[(size_t)t * D_MODEL + hh * DHEAD + tid];
  __syncthreads();

  float lmax = -1e30f;
  for (int jj = tid; jj < cnt; jj += 64) {
    const float* kr = k + (size_t)(b * SEQ + js + jj) * D_MODEL + hh * DHEAD;
    float dot = 0.f;
    for (int d = 0; d < DHEAD; ++d) dot += qs[d] * kr[d];
    dot *= 0.125f;                // 1/sqrt(64)
    sc[jj] = dot;
    lmax = fmaxf(lmax, dot);
  }
  red[tid] = lmax;
  __syncthreads();
  for (int off = 32; off > 0; off >>= 1) {
    if (tid < off) red[tid] = fmaxf(red[tid], red[tid + off]);
    __syncthreads();
  }
  float mx = red[0];
  __syncthreads();

  float lsum = 0.f;
  for (int jj = tid; jj < cnt; jj += 64) {
    float e = __expf(sc[jj] - mx);
    sc[jj] = e; lsum += e;
  }
  red[tid] = lsum;
  __syncthreads();
  for (int off = 32; off > 0; off >>= 1) {
    if (tid < off) red[tid] += red[tid + off];
    __syncthreads();
  }
  float inv = 1.f / red[0];

  float accd = 0.f;                  // tid == output dim
  for (int jj = 0; jj < cnt; ++jj)
    accd += sc[jj] * v[(size_t)(b * SEQ + js + jj) * D_MODEL + hh * DHEAD + tid];
  ctx[(size_t)t * D_MODEL + hh * DHEAD + tid] = accd * inv;
}

// ---------------- MoE router: logits -> top-2 softmax gates ----------------
__global__ __launch_bounds__(256)
void router_kernel(const float* __restrict__ hn, const float* __restrict__ rw,
                   const float* __restrict__ rb, float* __restrict__ gates)
{
  __shared__ float red[NEXP][256];
  int t = blockIdx.x, tid = threadIdx.x;
  const float* row = hn + (size_t)t * D_MODEL;
  float acc[NEXP] = {};
  for (int d = tid; d < D_MODEL; d += 256) {
    float hv = row[d];
    #pragma unroll
    for (int e = 0; e < NEXP; ++e) acc[e] += hv * rw[d * NEXP + e];
  }
  #pragma unroll
  for (int e = 0; e < NEXP; ++e) red[e][tid] = acc[e];
  __syncthreads();
  for (int off = 128; off > 0; off >>= 1) {
    if (tid < off)
      #pragma unroll
      for (int e = 0; e < NEXP; ++e) red[e][tid] += red[e][tid + off];
    __syncthreads();
  }
  if (tid == 0) {
    float l[NEXP];
    #pragma unroll
    for (int e = 0; e < NEXP; ++e) l[e] = red[e][0] + rb[e];
    int i0 = 0;
    for (int e = 1; e < NEXP; ++e) if (l[e] > l[i0]) i0 = e;
    int i1 = -1;
    for (int e = 0; e < NEXP; ++e) if (e != i0 && (i1 < 0 || l[e] > l[i1])) i1 = e;
    float e1 = __expf(l[i1] - l[i0]);
    float w0 = 1.f / (1.f + e1), w1 = e1 / (1.f + e1);
    #pragma unroll
    for (int e = 0; e < NEXP; ++e)
      gates[(size_t)t * NEXP + e] = (e == i0) ? w0 : (e == i1) ? w1 : 0.f;
  }
}

// ---------------- host orchestration ----------------
extern "C" void kernel_launch(void* const* d_in, const int* in_sizes, int n_in,
                              void* d_out, int out_size, void* d_ws, size_t ws_size,
                              hipStream_t stream)
{
  (void)in_sizes; (void)n_in; (void)out_size;

  // Workspace: h | hn | big(q,k,v,ctx <-> ffn-mid) | gates
  const size_t need = ((size_t)NTOK * D_MODEL * 2 + (size_t)NTOK * FFN_DIM
                       + (size_t)NTOK * NEXP) * sizeof(float);
  if (ws_size < need) return;

  auto F = [&](int i) { return (const float*)d_in[i]; };
  const int* ids = (const int*)d_in[0];

  struct Layer {
    const float *bk,*bo,*bq,*bv,*b1,*b2,*rb,*rw,*w1,*w2;
    const float *ln1b,*ln1s,*ln2b,*ln2s,*wk,*wo,*wq,*wv;
    bool moe;
  } L[4];

  // pytree leaf order (sorted dict keys): per block
  //   bk,bo,bq,bv, ffn{b1,b2,(rb,rw),w1,w2}, ln1_b,ln1_s,ln2_b,ln2_s, wk,wo,wq,wv
  int base = 1;
  for (int i = 0; i < 4; ++i) {
    bool moe = (i == 1) || (i == 3);
    L[i].moe = moe;
    L[i].bk = F(base + 0); L[i].bo = F(base + 1);
    L[i].bq = F(base + 2); L[i].bv = F(base + 3);
    L[i].b1 = F(base + 4); L[i].b2 = F(base + 5);
    int off;
    if (moe) { L[i].rb = F(base + 6); L[i].rw = F(base + 7);
               L[i].w1 = F(base + 8); L[i].w2 = F(base + 9); off = 10; }
    else     { L[i].rb = nullptr;     L[i].rw = nullptr;
               L[i].w1 = F(base + 6); L[i].w2 = F(base + 7); off = 8; }
    L[i].ln1b = F(base + off + 0); L[i].ln1s = F(base + off + 1);
    L[i].ln2b = F(base + off + 2); L[i].ln2s = F(base + off + 3);
    L[i].wk   = F(base + off + 4); L[i].wo   = F(base + off + 5);
    L[i].wq   = F(base + off + 6); L[i].wv   = F(base + off + 7);
    base += moe ? 18 : 16;
  }
  const float* emb  = F(base);       // embed  [V,D]
  const float* outw = F(base + 1);   // out_w  [V,D]

  float* ws    = (float*)d_ws;
  float* h     = ws;
  float* hn    = h  + (size_t)NTOK * D_MODEL;
  float* big   = hn + (size_t)NTOK * D_MODEL;        // NTOK*FFN_DIM floats
  float* gates = big + (size_t)NTOK * FFN_DIM;
  float* qb  = big;
  float* kb  = big + 1 * (size_t)NTOK * D_MODEL;
  float* vb  = big + 2 * (size_t)NTOK * D_MODEL;
  float* ctx = big + 3 * (size_t)NTOK * D_MODEL;

  auto gemm = [&](const float* A, const float* W, int ldw,
                  const float* bias, const float* res,
                  const float* rs, int rss,
                  float* C, int N, int K, int mode) {
    dim3 grid(N / 64, NTOK / 256);
    gemm_bf16_wmma<0><<<grid, 256, 0, stream>>>(A, K, W, ldw, bias, res,
                                                rs, rss, C, N, NTOK, N, K, mode);
  };

  embed_kernel<<<NTOK, 256, 0, stream>>>(ids, emb, h);

  for (int i = 0; i < 4; ++i) {
    const Layer& lp = L[i];
    // attention
    layernorm_kernel<<<NTOK, 256, 0, stream>>>(h, lp.ln1s, lp.ln1b, hn);
    gemm(hn, lp.wq, D_MODEL, lp.bq, nullptr, nullptr, 0, qb, D_MODEL, D_MODEL, GEMM_PLAIN);
    gemm(hn, lp.wk, D_MODEL, lp.bk, nullptr, nullptr, 0, kb, D_MODEL, D_MODEL, GEMM_PLAIN);
    gemm(hn, lp.wv, D_MODEL, lp.bv, nullptr, nullptr, 0, vb, D_MODEL, D_MODEL, GEMM_PLAIN);
    attention_kernel<<<dim3(SEQ, BATCH * NHEAD), 64, 0, stream>>>(qb, kb, vb, ctx);
    gemm(ctx, lp.wo, D_MODEL, lp.bo, h, nullptr, 0, h, D_MODEL, D_MODEL, GEMM_RES);
    // ffn
    layernorm_kernel<<<NTOK, 256, 0, stream>>>(h, lp.ln2s, lp.ln2b, hn);
    if (!lp.moe) {
      gemm(hn,  lp.w1, FFN_DIM, lp.b1, nullptr, nullptr, 0, big, FFN_DIM, D_MODEL, GEMM_RELU);
      gemm(big, lp.w2, D_MODEL, lp.b2, h, nullptr, 0, h, D_MODEL, FFN_DIM, GEMM_RES);
    } else {
      router_kernel<<<NTOK, 256, 0, stream>>>(hn, lp.rw, lp.rb, gates);
      for (int e = 0; e < NEXP; ++e) {
        gemm(hn,  lp.w1 + (size_t)e * D_MODEL * FFN_DIM, FFN_DIM,
             lp.b1 + (size_t)e * FFN_DIM, nullptr, nullptr, 0,
             big, FFN_DIM, D_MODEL, GEMM_RELU);
        gemm(big, lp.w2 + (size_t)e * FFN_DIM * D_MODEL, D_MODEL,
             lp.b2 + (size_t)e * D_MODEL, nullptr, gates + e, NEXP,
             h, D_MODEL, FFN_DIM, GEMM_SCALE_ACC);
      }
    }
  }

  // logits = h @ out_w^T  (transposed weight addressing)
  {
    dim3 grid(VOCAB / 64, NTOK / 256);
    gemm_bf16_wmma<1><<<grid, 256, 0, stream>>>(h, D_MODEL, outw, D_MODEL,
                                                nullptr, nullptr, nullptr, 0,
                                                (float*)d_out, VOCAB, NTOK,
                                                VOCAB, D_MODEL, GEMM_PLAIN);
  }
}
